// TensplitGAT_83081847374531
// MI455X (gfx1250) — compile-verified, matrix operand order checked by reading, not run on hardware
//
#include <hip/hip_runtime.h>
#include <hip/hip_bf16.h>
#include <stdint.h>

#define N_NODES_C 50000
#define N_EDGES_C 1600000
#define IN_DIM_C  256
#define HIDDEN_C  64
#define OUT_DIM_C 32
#define NEG_SLOPE_C 0.01f

typedef __attribute__((ext_vector_type(16))) __bf16 v16bf;
typedef __attribute__((ext_vector_type(8)))  float  v8f;

struct U8 { uint4 lo, hi; };

// ---- helpers -------------------------------------------------------------

__device__ __forceinline__ v16bf load_cvt_a(const float* p) {
  // 16 consecutive fp32 -> 16 bf16 (A-fragment half-row, 64B aligned)
  const float4* q = reinterpret_cast<const float4*>(p);
  float4 a = q[0], b = q[1], c = q[2], d = q[3];
  v16bf r;
  r[0]=(__bf16)a.x;  r[1]=(__bf16)a.y;  r[2]=(__bf16)a.z;  r[3]=(__bf16)a.w;
  r[4]=(__bf16)b.x;  r[5]=(__bf16)b.y;  r[6]=(__bf16)b.z;  r[7]=(__bf16)b.w;
  r[8]=(__bf16)c.x;  r[9]=(__bf16)c.y;  r[10]=(__bf16)c.z; r[11]=(__bf16)c.w;
  r[12]=(__bf16)d.x; r[13]=(__bf16)d.y; r[14]=(__bf16)d.z; r[15]=(__bf16)d.w;
  return r;
}

__device__ __forceinline__ v16bf load_b16(const uint16_t* p) {
  // 16 consecutive bf16 (32B aligned) straight into a B-fragment
  const uint4* q = reinterpret_cast<const uint4*>(p);
  U8 u; u.lo = q[0]; u.hi = q[1];
  return __builtin_bit_cast(v16bf, u);
}

__device__ __forceinline__ unsigned fenc(float f) {
  unsigned u = __float_as_uint(f);
  return (u & 0x80000000u) ? ~u : (u | 0x80000000u);
}
__device__ __forceinline__ float fdec(unsigned k) {
  unsigned u = (k & 0x80000000u) ? (k & 0x7fffffffu) : ~k;
  return __uint_as_float(u);
}

// ---- 1. weight transpose + bf16 convert ---------------------------------

__global__ void prep_weights(const float* __restrict__ W0, const float* __restrict__ W1,
                             uint16_t* __restrict__ W0T, uint16_t* __restrict__ W1T) {
  int i = blockIdx.x * 256 + threadIdx.x;
  if (i < IN_DIM_C * HIDDEN_C) {
    int k = i / HIDDEN_C, n = i % HIDDEN_C;
    __bf16 v = (__bf16)W0[i];
    W0T[n * IN_DIM_C + k] = __builtin_bit_cast(uint16_t, v);
  } else {
    int j = i - IN_DIM_C * HIDDEN_C;
    if (j < HIDDEN_C * OUT_DIM_C) {
      int k = j / OUT_DIM_C, n = j % OUT_DIM_C;
      __bf16 v = (__bf16)W1[j];
      W1T[n * HIDDEN_C + k] = __builtin_bit_cast(uint16_t, v);
    }
  }
}

// ---- 2. GEMM1: H0 = relu(X @ W0)  [50000,256]x[256,64] ------------------
// block = 128 threads (4 waves); blockIdx.x = 16-row tile; wave w -> cols w*16..w*16+15

__global__ void gemm1_relu(const float* __restrict__ X, const uint16_t* __restrict__ BT,
                           float* __restrict__ H0) {
  const int m0   = blockIdx.x * 16;
  const int lane = threadIdx.x & 31;
  const int wave = threadIdx.x >> 5;
  const int col0 = wave * 16;
  const int sub  = lane & 15;        // row (A) / col (B) within tile
  const int half = lane >> 4;        // K-half select (0: K 0..15, 1: K 16..31)

  const float*    arow = X  + (size_t)(m0  + sub) * IN_DIM_C + half * 16;
  const uint16_t* brow = BT + (size_t)(col0 + sub) * IN_DIM_C + half * 16;

  v8f c = {};
  #pragma unroll
  for (int k0 = 0; k0 < IN_DIM_C; k0 += 32) {
    v16bf a = load_cvt_a(arow + k0);
    v16bf b = load_b16(brow + k0);
    c = __builtin_amdgcn_wmma_f32_16x16x32_bf16(false, a, false, b, (short)0, c,
                                                false, false);
  }

  float* orow = H0 + (size_t)(m0 + half * 8) * HIDDEN_C + col0 + sub;
  #pragma unroll
  for (int r = 0; r < 8; ++r) {
    float v = c[r];
    orow[(size_t)r * HIDDEN_C] = v > 0.f ? v : 0.f;
  }
}

// ---- 3. GEMM2: H = H0 @ W1  [50000,64]x[64,32] --------------------------
// block = 64 threads (2 waves); blockIdx.x = 16-row tile

__global__ void gemm2(const float* __restrict__ H0, const uint16_t* __restrict__ BT,
                      float* __restrict__ H) {
  const int m0   = blockIdx.x * 16;
  const int lane = threadIdx.x & 31;
  const int wave = threadIdx.x >> 5;
  const int col0 = wave * 16;
  const int sub  = lane & 15;
  const int half = lane >> 4;

  const float*    arow = H0 + (size_t)(m0  + sub) * HIDDEN_C + half * 16;
  const uint16_t* brow = BT + (size_t)(col0 + sub) * HIDDEN_C + half * 16;

  v8f c = {};
  #pragma unroll
  for (int k0 = 0; k0 < HIDDEN_C; k0 += 32) {
    v16bf a = load_cvt_a(arow + k0);
    v16bf b = load_b16(brow + k0);
    c = __builtin_amdgcn_wmma_f32_16x16x32_bf16(false, a, false, b, (short)0, c,
                                                false, false);
  }

  float* orow = H + (size_t)(m0 + half * 8) * OUT_DIM_C + col0 + sub;
  #pragma unroll
  for (int r = 0; r < 8; ++r)
    orow[(size_t)r * OUT_DIM_C] = c[r];
}

// ---- 4. per-node attention projections ----------------------------------
// Padded column (index 32) of h is identically zero -> only att[0..31] and
// att[33..64] contribute.

__global__ void proj_s(const float* __restrict__ H, const float* __restrict__ att,
                       float* __restrict__ ssrc, float* __restrict__ sdst) {
  int i = blockIdx.x * 256 + threadIdx.x;
  if (i >= N_NODES_C) return;
  const float* h = H + (size_t)i * OUT_DIM_C;
  float a = 0.f, b = 0.f;
  #pragma unroll
  for (int d = 0; d < OUT_DIM_C; ++d) {
    float v = h[d];
    a += v * att[d];
    b += v * att[(OUT_DIM_C + 1) + d];
  }
  ssrc[i] = a;
  sdst[i] = b;
}

// ---- 5. zero-init accumulators (ws + d_out are poisoned) ----------------

__global__ void init_zero(float* __restrict__ H1, float* __restrict__ out,
                          float* __restrict__ denom, unsigned* __restrict__ mkey) {
  int i = blockIdx.x * 256 + threadIdx.x;
  if (i < N_NODES_C * OUT_DIM_C) { H1[i] = 0.f; out[i] = 0.f; }
  if (i < N_NODES_C)             { denom[i] = 0.f; mkey[i] = 0u; }
}

// ---- 6. edge pass 1: leaky_relu + segment max ---------------------------

__global__ void edge_max(const int* __restrict__ es, const int* __restrict__ ed,
                         const float* __restrict__ ssrc, const float* __restrict__ sdst,
                         float* __restrict__ ebuf, unsigned* __restrict__ mkey) {
  int i = blockIdx.x * 256 + threadIdx.x;
  if (i >= N_EDGES_C) return;
  int s = es[i], t = ed[i];
  float e = ssrc[s] + sdst[t];
  e = e > 0.f ? e : NEG_SLOPE_C * e;
  ebuf[i] = e;
  atomicMax(&mkey[s], fenc(e));
}

// ---- 7. edge pass 2: exp + segment sum ----------------------------------

__global__ void edge_exp(const int* __restrict__ es, float* __restrict__ ebuf,
                         const unsigned* __restrict__ mkey, float* __restrict__ denom) {
  int i = blockIdx.x * 256 + threadIdx.x;
  if (i >= N_EDGES_C) return;
  int s = es[i];
  float x = __expf(ebuf[i] - fdec(mkey[s]));
  ebuf[i] = x;
  atomicAdd(&denom[s], x);
}

// ---- 8/9. SpMM: out[src] += alpha * in[dst]  (one wave per edge) --------

__global__ void spmm(const int* __restrict__ es, const int* __restrict__ ed,
                     const float* __restrict__ ebuf, const float* __restrict__ denom,
                     const float* __restrict__ hin, float* __restrict__ hout) {
  long long idx = (long long)blockIdx.x * 256 + threadIdx.x;
  if (idx >= (long long)N_EDGES_C * OUT_DIM_C) return;
  int e = (int)(idx >> 5);
  int d = (int)(idx & 31);
  int s = es[e], t = ed[e];
  float alpha = ebuf[e] / denom[s];
  atomicAdd(&hout[(size_t)s * OUT_DIM_C + d], alpha * hin[(size_t)t * OUT_DIM_C + d]);
}

// ---- launch --------------------------------------------------------------

extern "C" void kernel_launch(void* const* d_in, const int* in_sizes, int n_in,
                              void* d_out, int out_size, void* d_ws, size_t ws_size,
                              hipStream_t stream) {
  const float* X   = (const float*)d_in[0];
  const float* W0  = (const float*)d_in[1];
  const float* W1  = (const float*)d_in[2];
  const float* att = (const float*)d_in[3];
  const int*   es  = (const int*)d_in[4];
  const int*   ed  = (const int*)d_in[5];
  float* out = (float*)d_out;

  char* ws = (char*)d_ws;
  // all sizes are multiples of 32 bytes -> uint4/float4 alignment preserved
  size_t oH0  = 0;
  size_t oH   = oH0  + (size_t)N_NODES_C * HIDDEN_C  * 4;
  size_t oH1  = oH   + (size_t)N_NODES_C * OUT_DIM_C * 4;
  size_t oSs  = oH1  + (size_t)N_NODES_C * OUT_DIM_C * 4;
  size_t oSd  = oSs  + (size_t)N_NODES_C * 4;
  size_t oDen = oSd  + (size_t)N_NODES_C * 4;
  size_t oMk  = oDen + (size_t)N_NODES_C * 4;
  size_t oEb  = oMk  + (size_t)N_NODES_C * 4;
  size_t oW0T = oEb  + (size_t)N_EDGES_C * 4;
  size_t oW1T = oW0T + (size_t)IN_DIM_C * HIDDEN_C * 2;

  float*    H0  = (float*)(ws + oH0);
  float*    H   = (float*)(ws + oH);
  float*    H1  = (float*)(ws + oH1);
  float*    Ss  = (float*)(ws + oSs);
  float*    Sd  = (float*)(ws + oSd);
  float*    Den = (float*)(ws + oDen);
  unsigned* Mk  = (unsigned*)(ws + oMk);
  float*    Eb  = (float*)(ws + oEb);
  uint16_t* W0T = (uint16_t*)(ws + oW0T);
  uint16_t* W1T = (uint16_t*)(ws + oW1T);

  const int wblk = (IN_DIM_C * HIDDEN_C + HIDDEN_C * OUT_DIM_C + 255) / 256;
  prep_weights<<<wblk, 256, 0, stream>>>(W0, W1, W0T, W1T);

  gemm1_relu<<<N_NODES_C / 16, 128, 0, stream>>>(X, W0T, H0);
  gemm2     <<<N_NODES_C / 16,  64, 0, stream>>>(H0, W1T, H);

  proj_s<<<(N_NODES_C + 255) / 256, 256, 0, stream>>>(H, att, Ss, Sd);

  init_zero<<<(N_NODES_C * OUT_DIM_C + 255) / 256, 256, 0, stream>>>(H1, out, Den, Mk);

  edge_max<<<(N_EDGES_C + 255) / 256, 256, 0, stream>>>(es, ed, Ss, Sd, Eb, Mk);
  edge_exp<<<(N_EDGES_C + 255) / 256, 256, 0, stream>>>(es, Eb, Mk, Den);

  const long long tot = (long long)N_EDGES_C * OUT_DIM_C;
  const int sblk = (int)((tot + 255) / 256);
  spmm<<<sblk, 256, 0, stream>>>(es, ed, Eb, Den, H,  H1);   // layer 1
  spmm<<<sblk, 256, 0, stream>>>(es, ed, Eb, Den, H1, out);  // layer 2 -> d_out
}
